// SepConv_27092653703316
// MI455X (gfx1250) — compile-verified, matrix operand order checked by reading, not run on hardware
//
#include <hip/hip_runtime.h>

typedef __attribute__((ext_vector_type(16))) _Float16 v16h;
typedef __attribute__((ext_vector_type(8)))  _Float16 v8h;
typedef __attribute__((ext_vector_type(8)))  float    v8f;

#define T_    4
#define B_    16
#define C_    256
#define H_    32
#define W_    32
#define HW_   1024
#define MED_  512
#define TB_   64
#define HP_   34
#define WP_   34
#define HWP_  1156
#define EPS_  1e-5f

// P (float) layout in workspace
#define P_INV1   0
#define P_SHF1   512
#define P_INV2   1024
#define P_SHF2   1536
#define P_INVP   2048
#define P_SHFP   2560
#define P_INV3   3072
#define P_SHF3   3328
#define P_BIAS0  3584
#define P_FLOATS 3840

// ---------------- prep kernels ----------------

__global__ void prep_bn_kernel(const float* g1, const float* b1, const float* m1, const float* v1,
                               const float* g2, const float* b2, const float* m2, const float* v2,
                               const float* gp, const float* bp, const float* mp, const float* vp,
                               const float* g3, const float* b3, const float* m3, const float* v3,
                               float* P) {
    int c = threadIdx.x;
    if (c < MED_) {
        float i1 = g1[c] * rsqrtf(v1[c] + EPS_);
        P[P_INV1 + c] = i1;  P[P_SHF1 + c] = b1[c] - m1[c] * i1;
        float i2 = g2[c] * rsqrtf(v2[c] + EPS_);
        P[P_INV2 + c] = i2;  P[P_SHF2 + c] = b2[c] - m2[c] * i2;
        float ip = gp[c] * rsqrtf(vp[c] + EPS_);
        P[P_INVP + c] = ip;  P[P_SHFP + c] = bp[c] - mp[c] * ip;
    }
    if (c < C_) {
        float i3 = g3[c] * rsqrtf(v3[c] + EPS_);
        P[P_INV3 + c] = i3;  P[P_SHF3 + c] = b3[c] - m3[c] * i3;
    }
}

__global__ void f32_to_f16_kernel(const float* __restrict__ w, _Float16* __restrict__ wh, int n) {
    int i = blockIdx.x * blockDim.x + threadIdx.x;
    if (i < n) wh[i] = (_Float16)w[i];
}

// wr1h[o,k] = w_rep1[o,k] * invp[k];  bias0[o] = sum_k shiftp[k] * w_rep1[o,k]
__global__ void prep_wr1_kernel(const float* __restrict__ w_rep1, const float* __restrict__ P,
                                _Float16* __restrict__ wrh, float* __restrict__ bias0) {
    int o = threadIdx.x;
    if (o >= C_) return;
    float acc = 0.f;
    for (int k = 0; k < MED_; ++k) {
        float w = w_rep1[o * MED_ + k];
        wrh[o * MED_ + k] = (_Float16)(w * P[P_INVP + k]);
        acc += w * P[P_SHFP + k];
    }
    bias0[o] = acc;
}

// ---------------- stage 1: LIF on x (NCHW) -> channel-last fp16 spikes ----------------
__global__ void lif1_kernel(const float* __restrict__ x, _Float16* __restrict__ s1) {
    __shared__ _Float16 tile[32][33];
    int tid = threadIdx.x;
    int lx = tid & 31;
    int ly = tid >> 5;
    int hw0 = blockIdx.x * 32;
    int c0  = blockIdx.y * 32;
    int b   = blockIdx.z;
    float mem[4] = {0.f, 0.f, 0.f, 0.f};
    float spk[4] = {0.f, 0.f, 0.f, 0.f};
    for (int t = 0; t < T_; ++t) {
        int tb = t * B_ + b;
        #pragma unroll
        for (int r = 0; r < 4; ++r) {
            int c = c0 + ly + 8 * r;
            float xi = x[((size_t)tb * C_ + c) * HW_ + hw0 + lx];
            mem[r] = (mem[r] - spk[r] * 0.5f) * 0.25f + xi;
            spk[r] = rintf(fminf(fmaxf(mem[r], 0.f), 1.f));
            tile[ly + 8 * r][lx] = (_Float16)spk[r];
        }
        __syncthreads();
        #pragma unroll
        for (int r = 0; r < 4; ++r) {
            int hw = hw0 + ly + 8 * r;
            s1[((size_t)tb * HW_ + hw) * C_ + c0 + lx] = tile[lx][ly + 8 * r];
        }
        __syncthreads();
    }
}

// ---------------- WMMA GEMM: Out[M,N] = A[M,K] @ Bw[N,K]^T, Out = acc*scale[n] + bias[n] ----
// Block 256 = 8 waves; block tile 128m x 64n; full-K B tile staged in LDS with an XOR
// bank swizzle (16B-chunk index ^ (row&15)) so each 16-lane fragment read hits 16
// distinct bank groups. A stream from global is software-pipelined one K-step ahead.
// NOTE: the A pipeline deliberately over-reads <=64B past the final K-step; A buffers
// are followed by larger workspace regions, so the touch stays in-bounds of d_ws.
__global__ void __launch_bounds__(256) gemm_wmma_kernel(const _Float16* __restrict__ A,
                                                        const _Float16* __restrict__ Bw,
                                                        float* __restrict__ Out,
                                                        const float* __restrict__ scale,
                                                        const float* __restrict__ bias,
                                                        int M, int N, int K) {
    extern __shared__ _Float16 lb[];          // [64][K] halves, xor-swizzled in 16B chunks
    const int vpr = K >> 3;                   // v8h vectors per row
    int tid  = threadIdx.x;
    int lane = tid & 31;
    int wave = tid >> 5;
    int m0 = blockIdx.x * 128 + wave * 16;
    int n0 = blockIdx.y * 64;

    // cooperative fill of the B tile (rows n0..n0+63, full K)
    {
        v8h* lbv = (v8h*)lb;
        int total = 64 * vpr;                 // 2048 (K=256) or 4096 (K=512)
        for (int idx = tid; idx < total; idx += 256) {
            int r  = idx / vpr;
            int vr = idx - r * vpr;
            v8h v = *(const v8h*)(Bw + (size_t)(n0 + r) * K + (vr << 3));
            lbv[r * vpr + (vr ^ (r & 15))] = v;
        }
    }
    __syncthreads();

    int nl   = lane & 15;
    int hi   = lane >> 4;
    int koff = hi * 8;                        // K sub-offset per the 16-bit A/B VGPR layout
    const _Float16* ap = A + (size_t)(m0 + nl) * K + koff;
    const v8h* lbv = (const v8h*)lb;

    v8f acc[4] = {};
    v8h alo = *(const v8h*)ap;                // K +0..7   (hi lanes: +8..15)
    v8h ahi = *(const v8h*)(ap + 16);         // K +16..23 (hi lanes: +24..31)
    for (int k0 = 0; k0 < K; k0 += 32) {
        // issue next A fragment (global) before consuming the current one
        v8h nlo = *(const v8h*)(ap + 32);
        v8h nhi = *(const v8h*)(ap + 48);
        __builtin_prefetch(ap + 64, 0, 1);
        v16h a;
        #pragma unroll
        for (int i = 0; i < 8; ++i) { a[i] = alo[i]; a[8 + i] = ahi[i]; }
        int vbase = (k0 + koff) >> 3;
        #pragma unroll
        for (int j = 0; j < 4; ++j) {
            int r  = j * 16 + nl;
            int sw = r & 15;
            v8h blo = lbv[(size_t)r * vpr + ((vbase)     ^ sw)];
            v8h bhi = lbv[(size_t)r * vpr + ((vbase + 2) ^ sw)];
            v16h bf;
            #pragma unroll
            for (int i = 0; i < 8; ++i) { bf[i] = blo[i]; bf[8 + i] = bhi[i]; }
            acc[j] = __builtin_amdgcn_wmma_f32_16x16x32_f16(
                false, a, false, bf, (short)0, acc[j], false, false);
        }
        alo = nlo; ahi = nhi;
        ap += 32;
    }
    // Epilogue: D layout -> element i of lane L is (M = i + 8*(L>>4), N = L&15)
    #pragma unroll
    for (int j = 0; j < 4; ++j) {
        int n = n0 + j * 16 + nl;
        float sc = scale ? scale[n] : 1.0f;
        float sh = bias[n];
        #pragma unroll
        for (int i = 0; i < 8; ++i) {
            int m = m0 + i + hi * 8;
            Out[(size_t)m * N + n] = acc[j][i] * sc + sh;
        }
    }
}

// ---------------- stage 2: LIF on channel-last y1 -> fp16 spikes ----------------
__global__ void lif2_kernel(const float* __restrict__ y1, _Float16* __restrict__ s2) {
    size_t g = (size_t)blockIdx.x * blockDim.x + threadIdx.x;
    if (g >= (size_t)B_ * HW_ * MED_) return;
    int c  = (int)(g % MED_);
    int hw = (int)((g / MED_) % HW_);
    int b  = (int)(g / ((size_t)MED_ * HW_));
    float mem = 0.f, spk = 0.f;
    for (int t = 0; t < T_; ++t) {
        size_t idx = ((size_t)(t * B_ + b) * HW_ + hw) * MED_ + c;
        mem = (mem - spk * 0.5f) * 0.25f + y1[idx];
        spk = rintf(fminf(fmaxf(mem, 0.f), 1.f));
        s2[idx] = (_Float16)spk;
    }
}

// depthwise 3x3 pad1 + BN2, channel-last
__global__ void dw2_kernel(const _Float16* __restrict__ s2, const float* __restrict__ wdw,
                           const float* __restrict__ P, float* __restrict__ y2) {
    size_t g = (size_t)blockIdx.x * blockDim.x + threadIdx.x;
    if (g >= (size_t)TB_ * HW_ * MED_) return;
    int c  = (int)(g % MED_);
    size_t rest = g / MED_;
    int hw = (int)(rest % HW_);
    int tb = (int)(rest / HW_);
    int h = hw >> 5, w = hw & 31;
    float acc = 0.f;
    #pragma unroll
    for (int dy = -1; dy <= 1; ++dy) {
        #pragma unroll
        for (int dx = -1; dx <= 1; ++dx) {
            int hh = h + dy, ww = w + dx;
            if ((unsigned)hh < (unsigned)H_ && (unsigned)ww < (unsigned)W_) {
                acc += (float)s2[((size_t)tb * HW_ + hh * W_ + ww) * MED_ + c]
                       * wdw[c * 9 + (dy + 1) * 3 + (dx + 1)];
            }
        }
    }
    y2[((size_t)tb * HW_ + hw) * MED_ + c] = acc * P[P_INV2 + c] + P[P_SHF2 + c];
}

__global__ void zero32_kernel(unsigned int* __restrict__ p, size_t n32) {
    size_t g = (size_t)blockIdx.x * blockDim.x + threadIdx.x;
    if (g < n32) p[g] = 0u;
}

// stage 3 LIF: spikes into zero-padded [TB, 34, 34, MED] fp16 (pad folded into bias0)
__global__ void lif3_kernel(const float* __restrict__ y2, _Float16* __restrict__ s3p) {
    size_t g = (size_t)blockIdx.x * blockDim.x + threadIdx.x;
    if (g >= (size_t)B_ * HW_ * MED_) return;
    int c  = (int)(g % MED_);
    int hw = (int)((g / MED_) % HW_);
    int b  = (int)(g / ((size_t)MED_ * HW_));
    int h = hw >> 5, w = hw & 31;
    float mem = 0.f, spk = 0.f;
    for (int t = 0; t < T_; ++t) {
        int tb = t * B_ + b;
        mem = (mem - spk * 0.5f) * 0.25f + y2[((size_t)tb * HW_ + hw) * MED_ + c];
        spk = rintf(fminf(fmaxf(mem, 0.f), 1.f));
        s3p[((size_t)tb * HWP_ + (h + 1) * WP_ + (w + 1)) * MED_ + c] = (_Float16)spk;
    }
}

// final depthwise 3x3 valid + BN3 -> output NCHW
__global__ void dw3_out_kernel(const float* __restrict__ r3, const float* __restrict__ wrep2,
                               const float* __restrict__ P, float* __restrict__ out) {
    unsigned int g = blockIdx.x * blockDim.x + threadIdx.x;
    if (g >= (unsigned)(TB_ * C_ * HW_)) return;
    int w  = g & 31;
    int h  = (g >> 5) & 31;
    int c  = (g >> 10) & 255;
    int tb = g >> 18;
    float acc = 0.f;
    #pragma unroll
    for (int i = 0; i < 3; ++i) {
        #pragma unroll
        for (int j = 0; j < 3; ++j) {
            acc += r3[((size_t)tb * HWP_ + (h + i) * WP_ + (w + j)) * C_ + c]
                   * wrep2[c * 9 + i * 3 + j];
        }
    }
    out[((size_t)tb * C_ + c) * HW_ + h * W_ + w] = acc * P[P_INV3 + c] + P[P_SHF3 + c];
}

// ---------------- host ----------------

extern "C" void kernel_launch(void* const* d_in, const int* in_sizes, int n_in,
                              void* d_out, int out_size, void* d_ws, size_t ws_size,
                              hipStream_t stream) {
    (void)in_sizes; (void)n_in; (void)out_size; (void)ws_size;
    const float* x      = (const float*)d_in[0];
    const float* w_pw1  = (const float*)d_in[1];
    const float* g1 = (const float*)d_in[2],  *b1 = (const float*)d_in[3];
    const float* m1 = (const float*)d_in[4],  *v1 = (const float*)d_in[5];
    const float* w_dw2  = (const float*)d_in[6];
    const float* g2 = (const float*)d_in[7],  *b2 = (const float*)d_in[8];
    const float* m2 = (const float*)d_in[9],  *v2 = (const float*)d_in[10];
    const float* gp = (const float*)d_in[11], *bp = (const float*)d_in[12];
    const float* mp = (const float*)d_in[13], *vp = (const float*)d_in[14];
    const float* w_rep1 = (const float*)d_in[15];
    const float* w_rep2 = (const float*)d_in[16];
    const float* g3 = (const float*)d_in[17], *b3 = (const float*)d_in[18];
    const float* m3 = (const float*)d_in[19], *v3 = (const float*)d_in[20];

    char* ws = (char*)d_ws;
    size_t off = 0;
    auto take = [&](size_t bytes) { size_t o = off; off = (off + bytes + 255) & ~(size_t)255; return o; };

    const int M1 = TB_ * HW_;        // 65536
    const int M3 = TB_ * HWP_;       // 73984 (multiple of 128)

    float*    P    = (float*)(ws + take(P_FLOATS * sizeof(float)));
    _Float16* w1h  = (_Float16*)(ws + take((size_t)MED_ * C_ * 2));
    _Float16* wrh  = (_Float16*)(ws + take((size_t)C_ * MED_ * 2));
    _Float16* s1   = (_Float16*)(ws + take((size_t)M1 * C_ * 2));
    float*    y12  = (float*)   (ws + take((size_t)M1 * MED_ * 4));   // y1, then reused for y2
    _Float16* s2   = (_Float16*)(ws + take((size_t)M1 * MED_ * 2));
    _Float16* s3p  = (_Float16*)(ws + take((size_t)M3 * MED_ * 2));
    float*    r3   = (float*)   (ws + take((size_t)M3 * C_ * 4));

    // prep
    prep_bn_kernel<<<1, 512, 0, stream>>>(g1, b1, m1, v1, g2, b2, m2, v2,
                                          gp, bp, mp, vp, g3, b3, m3, v3, P);
    f32_to_f16_kernel<<<(MED_ * C_ + 255) / 256, 256, 0, stream>>>(w_pw1, w1h, MED_ * C_);
    prep_wr1_kernel<<<1, 256, 0, stream>>>(w_rep1, P, wrh, P + P_BIAS0);

    // stage 1: LIF -> WMMA GEMM (M=65536, K=256, N=512), B tile in 32KB LDS
    lif1_kernel<<<dim3(HW_ / 32, C_ / 32, B_), 256, 0, stream>>>(x, s1);
    gemm_wmma_kernel<<<dim3(M1 / 128, MED_ / 64), 256, (size_t)64 * C_ * 2, stream>>>(
        s1, w1h, y12, P + P_INV1, P + P_SHF1, M1, MED_, C_);

    // stage 2
    lif2_kernel<<<(int)(((size_t)B_ * HW_ * MED_) / 256), 256, 0, stream>>>(y12, s2);
    dw2_kernel<<<(int)(((size_t)TB_ * HW_ * MED_) / 256), 256, 0, stream>>>(s2, w_dw2, P, y12);

    // stage 3: zero-padded spikes -> WMMA GEMM (M=73984, K=512, N=256), B tile in 64KB LDS
    size_t n32 = (size_t)M3 * MED_ / 2;
    zero32_kernel<<<(int)((n32 + 255) / 256), 256, 0, stream>>>((unsigned int*)s3p, n32);
    lif3_kernel<<<(int)(((size_t)B_ * HW_ * MED_) / 256), 256, 0, stream>>>(y12, s3p);
    gemm_wmma_kernel<<<dim3(M3 / 128, C_ / 64), 256, (size_t)64 * MED_ * 2, stream>>>(
        s3p, wrh, r3, nullptr, P + P_BIAS0, M3, C_, MED_);
    dw3_out_kernel<<<(TB_ * C_ * HW_) / 256, 256, 0, stream>>>(r3, w_rep2, P, (float*)d_out);
}
